// CosAttention_51874615001268
// MI455X (gfx1250) — matrix operand order repaired and checked by reading, not compiled, not run on hardware
//
#include <hip/hip_runtime.h>
#include <hip/hip_bf16.h>
#include <math.h>

// Problem constants (fixed by the reference):
// B=4, T=4096, D=1024, H=16, Dh=64  ->  M = B*T = 16384, K = 1024
#define MTOT 16384
#define DDIM 1024
#define HEADS 16
#define DH 64
#define TLEN 4096
#define NCHUNK 32      // chunks per (b,h) sequence
#define CLEN 128       // T per chunk (32*128 = 4096)

typedef __attribute__((ext_vector_type(16))) __bf16 v16bf;
typedef __attribute__((ext_vector_type(8)))  float  v8f;

// ---------- bf16 helpers (RNE) ----------
__device__ __forceinline__ unsigned short f2bf(float f) {
    union { float f; unsigned u; } x; x.f = f;
    unsigned u = x.u;
    u += 0x7FFFu + ((u >> 16) & 1u);
    return (unsigned short)(u >> 16);
}
__device__ __forceinline__ float bf2f(unsigned short h) {
    union { unsigned u; float f; } x; x.u = ((unsigned)h) << 16;
    return x.f;
}

// ---------- WMMA fragment loads from LDS ----------
// A tile in LDS: row-major [rows][32 k], 16-bit elems, row stride 32 elems.
// 16-bit A 16x32 layout: lane<16: row=lane, K=0..7 + 16..23 ; lane>=16: row=lane-16, K=8..15 + 24..31.
union FragU { uint4 u[2]; v16bf v; };

__device__ __forceinline__ v16bf load_fragA(const unsigned short* lA, int mrow0, int lane) {
    const int m   = mrow0 + (lane & 15);
    const int off = (lane >> 4) << 3;           // 0 or 8 elements
    const uint4* p = (const uint4*)(lA + m * 32 + off);
    FragU x;
    x.u[0] = p[0];      // K first 8
    x.u[1] = p[2];      // +16 elements (=32B) -> K second 8
    return x.v;
}
// B tile in LDS transposed: [n][32 k], row stride 32 elems.
// 16-bit B 32x16 layout: lane<16: col=lane, K=0..7 + 16..23 ; lane>=16: col=lane-16, K=8..15 + 24..31.
__device__ __forceinline__ v16bf load_fragB(const unsigned short* lB, int nrow0, int lane) {
    const int n   = nrow0 + (lane & 15);
    const int off = (lane >> 4) << 3;           // 0 or 8 elements
    const uint4* p = (const uint4*)(lB + n * 32 + off);
    FragU x;
    x.u[0] = p[0];
    x.u[1] = p[2];
    return x.v;
}

// ---------- Double-buffered bf16 WMMA GEMM ----------
// C[M,N] = A[M,1024] * Bt[N,1024]^T   (both operands pre-packed bf16, K contiguous)
// Block: 256 threads = 8 waves; block tile 128(M) x 128(N); wave tile 32x64 (8 WMMAs/step).
// CMODE: 0 = store bf16, 1 = store f32, 2 = store f32 nontemporal (final output)
// NGUARD: bounds-check N (only needed for the 32-wide q/k GEMM)
template<int CMODE, bool NGUARD>
__global__ __launch_bounds__(256) void gemm_bt_wmma(
    const unsigned short* __restrict__ Abf,   // [M,1024] bf16
    const unsigned short* __restrict__ Btg,   // [N,1024] bf16 (N-major)
    void* __restrict__ Cptr, int Ntot, int ldc)
{
    __shared__ unsigned short lA[2][128 * 32];   // 8 KB per buffer
    __shared__ unsigned short lB[2][128 * 32];   // 8 KB per buffer

    const int tid  = threadIdx.x;
    const int lane = tid & 31;
    const int wid  = tid >> 5;
    const int wm   = wid >> 1;                   // 0..3 : 32-row strip
    const int wn   = wid & 1;                    // 0..1 : 64-col strip
    const int m0   = blockIdx.y * 128;
    const int n0   = blockIdx.x * 128;

    // staging geometry: tile = 128 rows x 32 k = 512 uint4 (8 bf16) chunks; 2 per thread
    const int sr = tid >> 2;                     // 0..63 (+64 with q)
    const int sc = (tid & 3) << 3;               // element offset 0/8/16/24

    v8f acc[2][4] = {};
    uint4 ra[2], rb[2];

    // ---- stage K-tile 0 ----
    #pragma unroll
    for (int q = 0; q < 2; ++q) {
        const int r = sr + q * 64;
        ra[q] = *(const uint4*)(Abf + (size_t)(m0 + r) * DDIM + sc);
        if (NGUARD) {
            const int gn = n0 + r;
            rb[q] = (gn < Ntot) ? *(const uint4*)(Btg + (size_t)gn * DDIM + sc)
                                : make_uint4(0, 0, 0, 0);
        } else {
            rb[q] = *(const uint4*)(Btg + (size_t)(n0 + r) * DDIM + sc);
        }
    }
    #pragma unroll
    for (int q = 0; q < 2; ++q) {
        const int r = sr + q * 64;
        *(uint4*)(lA[0] + r * 32 + sc) = ra[q];
        *(uint4*)(lB[0] + r * 32 + sc) = rb[q];
    }
    __syncthreads();

    const int NT = DDIM / 32;   // 32 K-steps
    for (int kt = 0; kt < NT; ++kt) {
        const int cur = kt & 1;

        // ---- issue global loads for next K-tile (overlap with WMMA) ----
        if (kt + 1 < NT) {
            const int k0 = (kt + 1) * 32;
            #pragma unroll
            for (int q = 0; q < 2; ++q) {
                const int r = sr + q * 64;
                ra[q] = *(const uint4*)(Abf + (size_t)(m0 + r) * DDIM + k0 + sc);
                if (NGUARD) {
                    const int gn = n0 + r;
                    rb[q] = (gn < Ntot) ? *(const uint4*)(Btg + (size_t)gn * DDIM + k0 + sc)
                                        : make_uint4(0, 0, 0, 0);
                } else {
                    rb[q] = *(const uint4*)(Btg + (size_t)(n0 + r) * DDIM + k0 + sc);
                }
            }
            // deep prefetch (global_prefetch_b8 on gfx1250)
            if (kt + 2 < NT)
                __builtin_prefetch(Abf + (size_t)(m0 + (tid >> 1)) * DDIM + (kt + 2) * 32, 0, 1);
        }

        // ---- compute: wave tile 32x64 = 2x4 WMMA tiles ----
        v16bf a0 = load_fragA(lA[cur], wm * 32 + 0,  lane);
        v16bf a1 = load_fragA(lA[cur], wm * 32 + 16, lane);
        v16bf b0 = load_fragB(lB[cur], wn * 64 + 0,  lane);
        v16bf b1 = load_fragB(lB[cur], wn * 64 + 16, lane);
        v16bf b2 = load_fragB(lB[cur], wn * 64 + 32, lane);
        v16bf b3 = load_fragB(lB[cur], wn * 64 + 48, lane);

        acc[0][0] = __builtin_amdgcn_wmma_f32_16x16x32_bf16(false, a0, false, b0, (short)0, acc[0][0], false, false);
        acc[0][1] = __builtin_amdgcn_wmma_f32_16x16x32_bf16(false, a0, false, b1, (short)0, acc[0][1], false, false);
        acc[0][2] = __builtin_amdgcn_wmma_f32_16x16x32_bf16(false, a0, false, b2, (short)0, acc[0][2], false, false);
        acc[0][3] = __builtin_amdgcn_wmma_f32_16x16x32_bf16(false, a0, false, b3, (short)0, acc[0][3], false, false);
        acc[1][0] = __builtin_amdgcn_wmma_f32_16x16x32_bf16(false, a1, false, b0, (short)0, acc[1][0], false, false);
        acc[1][1] = __builtin_amdgcn_wmma_f32_16x16x32_bf16(false, a1, false, b1, (short)0, acc[1][1], false, false);
        acc[1][2] = __builtin_amdgcn_wmma_f32_16x16x32_bf16(false, a1, false, b2, (short)0, acc[1][2], false, false);
        acc[1][3] = __builtin_amdgcn_wmma_f32_16x16x32_bf16(false, a1, false, b3, (short)0, acc[1][3], false, false);

        // ---- store next tile into the other buffer (disjoint from current reads) ----
        if (kt + 1 < NT) {
            unsigned short* sA = lA[cur ^ 1];
            unsigned short* sB = lB[cur ^ 1];
            #pragma unroll
            for (int q = 0; q < 2; ++q) {
                const int r = sr + q * 64;
                *(uint4*)(sA + r * 32 + sc) = ra[q];
                *(uint4*)(sB + r * 32 + sc) = rb[q];
            }
        }
        __syncthreads();
    }

    // ---- store C: VGPR r -> row (lane>=16 ? 8 : 0)+r, col = lane&15 ----
    const int col = lane & 15;
    const int rbase = (lane >> 4) << 3;
    #pragma unroll
    for (int i = 0; i < 2; ++i) {
        #pragma unroll
        for (int j = 0; j < 4; ++j) {
            const int gm = m0 + wm * 32 + i * 16;
            const int gn = n0 + wn * 64 + j * 16 + col;
            if (!NGUARD || gn < Ntot) {
                #pragma unroll
                for (int r = 0; r < 8; ++r) {
                    const size_t idx = (size_t)(gm + rbase + r) * ldc + gn;
                    const float val = acc[i][j][r];
                    if (CMODE == 0)      ((unsigned short*)Cptr)[idx] = f2bf(val);
                    else if (CMODE == 1) ((float*)Cptr)[idx] = val;
                    else                 __builtin_nontemporal_store(val, &((float*)Cptr)[idx]);
                }
            }
        }
    }
}

// ---------- One-shot packing kernels ----------
// f32 -> bf16 bulk convert (query): n4 = elements/4
__global__ __launch_bounds__(256) void convert_bf16_kernel(
    const float* __restrict__ in, unsigned short* __restrict__ out, int n4)
{
    const int i = blockIdx.x * 256 + threadIdx.x;
    if (i < n4) {
        const float4 v = ((const float4*)in)[i];
        ((ushort4*)out)[i] = make_ushort4(f2bf(v.x), f2bf(v.y), f2bf(v.z), f2bf(v.w));
    }
}
// W_V [H,D,Dh] -> Bt[n=h*64+e][k=d]  (1024 x 1024)
__global__ __launch_bounds__(256) void pack_WV_kernel(
    const float* __restrict__ W, unsigned short* __restrict__ Bt)
{
    const int i = blockIdx.x * 256 + threadIdx.x;      // 1024*256 quads
    const int n = i >> 8, k4 = (i & 255) << 2;
    const int h = n >> 6, e = n & 63;
    const float* s = W + (size_t)h * (DDIM * DH) + (size_t)k4 * DH + e;   // k stride = 64
    ((ushort4*)(Bt + (size_t)n * DDIM))[k4 >> 2] =
        make_ushort4(f2bf(s[0]), f2bf(s[64]), f2bf(s[128]), f2bf(s[192]));
}
// W_Q/W_K [H,D,1] -> Bt[n][k] with n<16 from W_Q, n>=16 from W_K  (32 x 1024)
__global__ __launch_bounds__(256) void pack_WQK_kernel(
    const float* __restrict__ WQ, const float* __restrict__ WK, unsigned short* __restrict__ Bt)
{
    const int i = blockIdx.x * 256 + threadIdx.x;      // 32*256 quads
    const int n = i >> 8, k4 = (i & 255) << 2;
    const float* s = (n < 16) ? (WQ + (size_t)n * DDIM + k4)
                              : (WK + (size_t)(n - 16) * DDIM + k4);
    const float4 v = *(const float4*)s;
    ((ushort4*)(Bt + (size_t)n * DDIM))[k4 >> 2] =
        make_ushort4(f2bf(v.x), f2bf(v.y), f2bf(v.z), f2bf(v.w));
}
// W_O [D,Dh,H] -> Bt[n=d][k=h*64+e] = W_O[d*1024 + e*16 + h]  (1024 x 1024)
__global__ __launch_bounds__(256) void pack_WO_kernel(
    const float* __restrict__ W, unsigned short* __restrict__ Bt)
{
    const int i = blockIdx.x * 256 + threadIdx.x;      // 1024*256 quads
    const int n = i >> 8, k4 = (i & 255) << 2;
    const int h = k4 >> 6, e = k4 & 63;                // k4..k4+3 share h (k4 multiple of 4)
    const float* s = W + (size_t)n * DDIM + (size_t)e * HEADS + h;        // e stride = 16
    ((ushort4*)(Bt + (size_t)n * DDIM))[k4 >> 2] =
        make_ushort4(f2bf(s[0]), f2bf(s[16]), f2bf(s[32]), f2bf(s[48]));
}

// ---------- Elementwise: angles -> cos/sin of tanh(x)*pi/4 ----------
__global__ __launch_bounds__(256) void angles_kernel(
    const float* __restrict__ qk,       // [16384, 32] (q cols 0..15, k cols 16..31)
    float* __restrict__ cq, float* __restrict__ sq,
    float* __restrict__ ck, float* __restrict__ sk)
{
    const int i = blockIdx.x * 256 + threadIdx.x;   // 16384*32 total
    const int row = i >> 5;
    const int col = i & 31;
    const float a = tanhf(qk[i]) * 0.78539816339744831f;
    float s, c;
    sincosf(a, &s, &c);
    if (col < 16) { cq[row * 16 + col] = c; sq[row * 16 + col] = s; }
    else          { ck[row * 16 + col - 16] = c; sk[row * 16 + col - 16] = s; }
}

// ---------- Scan pass 1: per-chunk sums ----------
__global__ __launch_bounds__(64) void scan_pass1(
    const unsigned short* __restrict__ Vbf,
    const float* __restrict__ ck, const float* __restrict__ sk,
    float* __restrict__ pC, float* __restrict__ pS,
    float* __restrict__ pCk, float* __restrict__ pSk)
{
    const int blk = blockIdx.x;            // b*H*NCHUNK + h*NCHUNK + c
    const int c = blk & (NCHUNK - 1);
    const int h = (blk >> 5) & (HEADS - 1);
    const int b = blk >> 9;
    const int e = threadIdx.x;
    const int t0 = c * CLEN;
    float sc = 0.f, ss = 0.f, sck = 0.f, ssk = 0.f;
    for (int t = t0; t < t0 + CLEN; ++t) {
        const int row = b * TLEN + t;
        const float ckv = ck[row * HEADS + h];
        const float skv = sk[row * HEADS + h];
        const float v = bf2f(Vbf[(size_t)row * DDIM + h * DH + e]);
        sc += ckv * v; ss += skv * v;
        sck += ckv;    ssk += skv;
    }
    pC[(size_t)blk * 64 + e] = sc;
    pS[(size_t)blk * 64 + e] = ss;
    if (e == 0) { pCk[blk] = sck; pSk[blk] = ssk; }
}

// ---------- Scan pass 2: in-place exclusive scan of chunk totals ----------
__global__ __launch_bounds__(64) void scan_pass2(
    float* __restrict__ pC, float* __restrict__ pS,
    float* __restrict__ pCk, float* __restrict__ pSk)
{
    const int bh = blockIdx.x;      // 0..63
    const int e  = threadIdx.x;     // 0..63
    float rc = 0.f, rs = 0.f;
    for (int c = 0; c < NCHUNK; ++c) {
        const size_t idx = ((size_t)bh * NCHUNK + c) * 64 + e;
        const float tc = pC[idx]; pC[idx] = rc; rc += tc;
        const float ts = pS[idx]; pS[idx] = rs; rs += ts;
    }
    if (e == 0) {
        float r0 = 0.f, r1 = 0.f;
        for (int c = 0; c < NCHUNK; ++c) {
            const int idx = bh * NCHUNK + c;
            const float t0 = pCk[idx]; pCk[idx] = r0; r0 += t0;
            const float t1 = pSk[idx]; pSk[idx] = r1; r1 += t1;
        }
    }
}

// ---------- Scan pass 3: running prefix + normalized heads (bf16) ----------
__global__ __launch_bounds__(64) void scan_pass3(
    const unsigned short* __restrict__ Vbf,
    const float* __restrict__ cq, const float* __restrict__ sq,
    const float* __restrict__ ck, const float* __restrict__ sk,
    const float* __restrict__ pC, const float* __restrict__ pS,
    const float* __restrict__ pCk, const float* __restrict__ pSk,
    unsigned short* __restrict__ heads)
{
    const int blk = blockIdx.x;
    const int c = blk & (NCHUNK - 1);
    const int h = (blk >> 5) & (HEADS - 1);
    const int b = blk >> 9;
    const int e = threadIdx.x;
    const int t0 = c * CLEN;

    float runC  = pC[(size_t)blk * 64 + e];
    float runS  = pS[(size_t)blk * 64 + e];
    float runCk = pCk[blk];
    float runSk = pSk[blk];

    for (int t = t0; t < t0 + CLEN; ++t) {
        const int row = b * TLEN + t;
        const float ckv = ck[row * HEADS + h];
        const float skv = sk[row * HEADS + h];
        const float v = bf2f(Vbf[(size_t)row * DDIM + h * DH + e]);
        runC += ckv * v;  runS += skv * v;
        runCk += ckv;     runSk += skv;
        const float cqv = cq[row * HEADS + h];
        const float sqv = sq[row * HEADS + h];
        const float num = cqv * runC + sqv * runS;
        const float den = cqv * runCk + sqv * runSk + 1e-6f;
        heads[(size_t)row * DDIM + h * DH + e] = f2bf(num / den);
    }
}

// ---------- Host launcher ----------
extern "C" void kernel_launch(void* const* d_in, const int* in_sizes, int n_in,
                              void* d_out, int out_size, void* d_ws, size_t ws_size,
                              hipStream_t stream) {
    const float* query = (const float*)d_in[0];   // [4,4096,1024]
    const float* W_Q   = (const float*)d_in[1];   // [16,1024,1]
    const float* W_K   = (const float*)d_in[2];   // [16,1024,1]
    const float* W_V   = (const float*)d_in[3];   // [16,1024,64]
    const float* W_O   = (const float*)d_in[4];   // [1024,64,16]
    float* out = (float*)d_out;                   // [4,4096,1024]

    char* ws = (char*)d_ws;
    // Workspace layout (bytes); heads aliases Abf (dead after the V GEMM).
    unsigned short* Abf   = (unsigned short*)(ws);                 // 32 MB (query bf16)
    unsigned short* headsb= Abf;                                   // alias, written after Abf's last use
    unsigned short* Vbf   = (unsigned short*)(ws + (32u << 20));   // 32 MB
    unsigned short* BtV   = (unsigned short*)(ws + (64u << 20));   // 2 MB
    unsigned short* BtO   = (unsigned short*)(ws + (66u << 20));   // 2 MB
    unsigned short* BtQK  = (unsigned short*)(ws + (68u << 20));   // 64 KB
    float* qk  = (float*)(ws + (69u << 20));                       // 2 MB
    float* cq  = (float*)(ws + (71u << 20));                       // 1 MB each
    float* sq  = (float*)(ws + (72u << 20));
    float* ck  = (float*)(ws + (73u << 20));
    float* sk  = (float*)(ws + (74u << 20));
    float* pC  = (float*)(ws + (75u << 20));                       // 512 KB
    float* pS  = (float*)(ws + (75u << 20) + (512u << 10));        // 512 KB
    float* pCk = (float*)(ws + (76u << 20));                       // 8 KB
    float* pSk = (float*)(ws + (76u << 20) + (16u << 10));         // 8 KB

    // 0) pack operands to bf16 / GEMM-friendly layouts
    convert_bf16_kernel<<<(MTOT * DDIM / 4 + 255) / 256, 256, 0, stream>>>(query, Abf, MTOT * DDIM / 4);
    pack_WQK_kernel<<<32, 256, 0, stream>>>(W_Q, W_K, BtQK);
    pack_WV_kernel<<<1024, 256, 0, stream>>>(W_V, BtV);
    pack_WO_kernel<<<1024, 256, 0, stream>>>(W_O, BtO);

    // 1) q,k projections: [16384,1024] x [1024,32] -> f32 (guarded N)
    gemm_bt_wmma<1, true><<<dim3(1, MTOT / 128), 256, 0, stream>>>(Abf, BtQK, qk, 32, 32);

    // 2) angles -> cos/sin
    angles_kernel<<<(MTOT * 32) / 256, 256, 0, stream>>>(qk, cq, sq, ck, sk);

    // 3) V projection: [16384,1024] x [1024,1024] -> bf16 (unguarded)
    gemm_bt_wmma<0, false><<<dim3(DDIM / 128, MTOT / 128), 256, 0, stream>>>(Abf, BtV, Vbf, DDIM, DDIM);

    // 4) chunked causal scan
    scan_pass1<<<4 * HEADS * NCHUNK, 64, 0, stream>>>(Vbf, ck, sk, pC, pS, pCk, pSk);
    scan_pass2<<<4 * HEADS, 64, 0, stream>>>(pC, pS, pCk, pSk);
    scan_pass3<<<4 * HEADS * NCHUNK, 64, 0, stream>>>(
        Vbf, cq, sq, ck, sk, pC, pS, pCk, pSk, headsb);

    // 5) output projection: heads[16384,1024](bf16) x BtO -> f32 out (nontemporal)
    gemm_bt_wmma<2, false><<<dim3(DDIM / 128, MTOT / 128), 256, 0, stream>>>(headsb, BtO, out, DDIM, DDIM);
}
